// Attention_22660247454077
// MI455X (gfx1250) — matrix operand order repaired and checked by reading, not compiled, not run on hardware
//
#include <hip/hip_runtime.h>
#include <stdint.h>

#define B_     2
#define N_     320
#define DIM_   256
#define EDIM_  128
#define H_     8
#define DH_    64
#define INNER_ 512
#define JT     64
#define EDSF   (EDIM_ + 4)   /* 132: padded f32 LDS stride (4-bank row skew) */
#define SCALE  0.125f        /* DH^-0.5 */

typedef __bf16 v16bf __attribute__((ext_vector_type(16)));
typedef __bf16 v8bf  __attribute__((ext_vector_type(8)));
typedef float  v8f   __attribute__((ext_vector_type(8)));
typedef float  v4f   __attribute__((ext_vector_type(4)));

static __device__ __forceinline__ v16bf mk16(v8bf lo, v8bf hi) {
  v16bf r;
#pragma unroll
  for (int t = 0; t < 8; ++t) { r[t] = lo[t]; r[t + 8] = hi[t]; }
  return r;
}

// 16-byte async copy global -> LDS (gfx1250, tracked by ASYNCcnt)
static __device__ __forceinline__ void async_cp16(unsigned lds_addr, const float* g) {
  const unsigned long long ga = (unsigned long long)(uintptr_t)g;
  asm volatile("global_load_async_to_lds_b128 %0, %1, off"
               :: "v"(lds_addr), "v"(ga) : "memory");
}
#define WAIT_ASYNC_LE8() asm volatile("s_wait_asynccnt 0x8" ::: "memory")
#define WAIT_ASYNC_0()   asm volatile("s_wait_asynccnt 0x0" ::: "memory")

// ---------------------------------------------------------------------------
// Setup: We (128x512 f32) -> Wet (512x128 bf16) in workspace.  128 KB,
// L2-resident, shared by all attention blocks (B fragments become two
// contiguous 16B loads per lane instead of 16 strided scalars).
// ---------------------------------------------------------------------------
__global__ __launch_bounds__(256) void transpose_we(const float* __restrict__ We,
                                                    __bf16* __restrict__ Wet) {
  const int idx = blockIdx.x * 256 + threadIdx.x;   // grid 256 -> 65536 elems
  const int k = idx >> 9;          // EDIM row
  const int n = idx & 511;         // INNER col
  Wet[n * EDIM_ + k] = (__bf16)We[idx];
}

// ---------------------------------------------------------------------------
// Kernel 1: projections.  LDW is a template constant so B-fragment loads use
// immediate offsets (independent, clause-able).  TRANS=false: row-major Q.
// TRANS=true: per-head transposed [b][h][d][j] for K/V.
// grid (40, 8)  block 128 (4 waves, one 16x16 WMMA tile each)
// ---------------------------------------------------------------------------
template <int LDW, bool TRANS>
__global__ __launch_bounds__(128) void proj_gemm(
    const float* __restrict__ nodes, const float* __restrict__ W,
    const float* __restrict__ bias, int wcol_off, float* __restrict__ dst) {
  const int rb = blockIdx.x;
  const int cb = blockIdx.y;
  const int wave = threadIdx.x >> 5;
  const int lane = threadIdx.x & 31;
  const int L0 = lane & 15, g = lane >> 4;
  const int row0 = rb * 16;
  const int col0 = cb * 64 + wave * 16;
  const int wcol = col0 + wcol_off;

  v8f c = {0.f, 0.f, 0.f, 0.f, 0.f, 0.f, 0.f, 0.f};
#pragma unroll
  for (int k0 = 0; k0 < DIM_; k0 += 32) {
    const float* ap = nodes + (size_t)(row0 + L0) * DIM_ + k0 + 16 * g;
    v16bf a;
#pragma unroll
    for (int e = 0; e < 8; ++e) { a[e] = (__bf16)ap[e]; a[e + 8] = (__bf16)ap[8 + e]; }
    const float* bp = W + (size_t)(k0 + 16 * g) * LDW + wcol + L0;
    v16bf bb;
#pragma unroll
    for (int e = 0; e < 16; ++e) bb[e] = (__bf16)bp[e * LDW];   // const offsets
    c = __builtin_amdgcn_wmma_f32_16x16x32_bf16(false, a, false, bb, (short)0, c,
                                                false, false);
  }
  const float bv = bias[wcol + L0];
#pragma unroll
  for (int r = 0; r < 8; ++r) {
    const int row = row0 + r + 8 * g;
    const int col = col0 + L0;
    const float val = c[r] + bv;
    if (!TRANS) {
      dst[(size_t)row * INNER_ + col] = val;
    } else {
      const int bb_ = row / N_, j = row % N_;
      const int h = col / DH_, d = col % DH_;
      dst[(((size_t)bb_ * H_ + h) * DH_ + d) * N_ + j] = val;
    }
  }
}

// ---------------------------------------------------------------------------
// Kernel 2: fused edge-GEMM + flash attention.  One block per (b,i); wave =
// head.  Edge tiles (64x128 f32) double-buffered in LDS via async-to-LDS
// copies (prefetch next tile under current tile's 512 WMMAs).  E kept in
// accumulator VGPRs, consumed for scores (k+be+E) and output (v+be+E) with
// online softmax; all reductions via shfl_xor (no LDS round trips).
// ---------------------------------------------------------------------------
__global__ __launch_bounds__(256) void fused_attn(
    const float* __restrict__ edges, const __bf16* __restrict__ Wet,
    const float* __restrict__ be, const float* __restrict__ Q,
    const float* __restrict__ Kt, const float* __restrict__ Vt,
    float* __restrict__ O) {
  __shared__ __align__(16) float ed_lds[2][JT * EDSF];   // 2 x 33792 B

  const int bi = blockIdx.x;   // b*N + i
  const int b  = bi / N_;
  const int tid  = threadIdx.x;
  const int wave = tid >> 5;   // head
  const int lane = tid & 31;
  const int L0 = lane & 15, g = lane >> 4;

  // per-lane constants for this head: q, be at d = ni*16 + L0
  float q4[4], be4[4];
#pragma unroll
  for (int ni = 0; ni < 4; ++ni) {
    const int col = wave * DH_ + ni * 16 + L0;
    q4[ni]  = Q[(size_t)bi * INNER_ + col];
    be4[ni] = be[col];
  }

  float m_run = -1e30f, l_run = 0.f;
  float acc4[4] = {0.f, 0.f, 0.f, 0.f};
  const v8f zero8 = {0.f, 0.f, 0.f, 0.f, 0.f, 0.f, 0.f, 0.f};
  const size_t erow = (size_t)bi * N_ * EDIM_;

  // prologue: async-stage tile 0
#pragma unroll
  for (int it = 0; it < 8; ++it) {
    const int ch = it * 256 + tid;            // 2048 16B-chunks per tile
    const int jj = ch >> 5, cc = (ch & 31) * 4;
    async_cp16((unsigned)(uintptr_t)&ed_lds[0][jj * EDSF + cc],
               edges + erow + (size_t)jj * EDIM_ + cc);
  }

  for (int t = 0; t < N_ / JT; ++t) {
    const int j0 = t * JT;
    if (t + 1 < N_ / JT) {       // prefetch next tile into the other buffer
      const size_t nb = erow + (size_t)(j0 + JT) * EDIM_;
#pragma unroll
      for (int it = 0; it < 8; ++it) {
        const int ch = it * 256 + tid;
        const int jj = ch >> 5, cc = (ch & 31) * 4;
        async_cp16((unsigned)(uintptr_t)&ed_lds[(t + 1) & 1][jj * EDSF + cc],
                   edges + nb + (size_t)jj * EDIM_ + cc);
      }
      WAIT_ASYNC_LE8();          // in-order: current tile's 8 copies done
    } else {
      WAIT_ASYNC_0();
    }
    __syncthreads();
    const float* eb = ed_lds[t & 1];

    // --- E = edges_tile(64x128) @ We_head(128x64) via WMMA ---
    v8f eacc[4][4];
#pragma unroll
    for (int mi = 0; mi < 4; ++mi)
#pragma unroll
      for (int ni = 0; ni < 4; ++ni) eacc[mi][ni] = zero8;

#pragma unroll
    for (int k0 = 0; k0 < EDIM_; k0 += 32) {
      v16bf afr[4];
#pragma unroll
      for (int mi = 0; mi < 4; ++mi) {
        const int base = (mi * 16 + L0) * EDSF + k0 + 8 * g;
        const v4f x0 = *(const v4f*)&eb[base];
        const v4f x1 = *(const v4f*)&eb[base + 4];
        const v4f y0 = *(const v4f*)&eb[base + 16];
        const v4f y1 = *(const v4f*)&eb[base + 20];
        v16bf a;
#pragma unroll
        for (int u = 0; u < 4; ++u) {
          a[u]      = (__bf16)x0[u];
          a[u + 4]  = (__bf16)x1[u];
          a[u + 8]  = (__bf16)y0[u];
          a[u + 12] = (__bf16)y1[u];
        }
        afr[mi] = a;
      }
#pragma unroll
      for (int ni = 0; ni < 4; ++ni) {
        const __bf16* wp = Wet + (wave * DH_ + ni * 16 + L0) * EDIM_ + k0 + 8 * g;
        const v16bf bfr = mk16(*(const v8bf*)wp, *(const v8bf*)(wp + 16));
#pragma unroll
        for (int mi = 0; mi < 4; ++mi)
          eacc[mi][ni] = __builtin_amdgcn_wmma_f32_16x16x32_bf16(
              false, afr[mi], false, bfr, (short)0, eacc[mi][ni], false, false);
      }
    }

    // --- scores: s[j] = scale * q . (K[j] + be + E[j]) ---
    float sc[4][8];
#pragma unroll
    for (int mi = 0; mi < 4; ++mi) {
#pragma unroll
      for (int r = 0; r < 8; ++r) sc[mi][r] = 0.f;
#pragma unroll
      for (int ni = 0; ni < 4; ++ni) {
        const float* kp = Kt + (((size_t)(b * H_ + wave)) * DH_ + ni * 16 + L0) * N_ +
                          j0 + mi * 16 + 8 * g;
        const v4f ka = *(const v4f*)kp;
        const v4f kb = *(const v4f*)(kp + 4);
        const float qv = q4[ni], bev = be4[ni];
#pragma unroll
        for (int r = 0; r < 4; ++r) {
          sc[mi][r]     += qv * (ka[r] + bev + eacc[mi][ni][r]);
          sc[mi][r + 4] += qv * (kb[r] + bev + eacc[mi][ni][r + 4]);
        }
      }
#pragma unroll
      for (int r = 0; r < 8; ++r) {   // reduce over the 16 d-lanes of the group
        float v = sc[mi][r];
        v += __shfl_xor(v, 1);
        v += __shfl_xor(v, 2);
        v += __shfl_xor(v, 4);
        v += __shfl_xor(v, 8);
        sc[mi][r] = v * SCALE;
      }
    }

    // --- online softmax update ---
    float mt = -1e30f;
#pragma unroll
    for (int mi = 0; mi < 4; ++mi)
#pragma unroll
      for (int r = 0; r < 8; ++r) mt = fmaxf(mt, sc[mi][r]);
    mt = fmaxf(mt, __shfl_xor(mt, 16));
    const float m_new = fmaxf(m_run, mt);
    const float alpha = __expf(m_run - m_new);
    l_run *= alpha;
#pragma unroll
    for (int ni = 0; ni < 4; ++ni) acc4[ni] *= alpha;
    float lsum = 0.f;
#pragma unroll
    for (int mi = 0; mi < 4; ++mi)
#pragma unroll
      for (int r = 0; r < 8; ++r) {
        const float p = __expf(sc[mi][r] - m_new);
        sc[mi][r] = p;
        lsum += p;
      }
    l_run += lsum + __shfl_xor(lsum, 16);
    m_run = m_new;

    // --- out accumulation: acc[d] += p_j * (V[j] + be + E[j]) ---
#pragma unroll
    for (int mi = 0; mi < 4; ++mi)
#pragma unroll
      for (int ni = 0; ni < 4; ++ni) {
        const float* vp = Vt + (((size_t)(b * H_ + wave)) * DH_ + ni * 16 + L0) * N_ +
                          j0 + mi * 16 + 8 * g;
        const v4f va = *(const v4f*)vp;
        const v4f vb = *(const v4f*)(vp + 4);
        const float bev = be4[ni];
        float a = acc4[ni];
#pragma unroll
        for (int r = 0; r < 4; ++r) {
          a += sc[mi][r]     * (va[r] + bev + eacc[mi][ni][r]);
          a += sc[mi][r + 4] * (vb[r] + bev + eacc[mi][ni][r + 4]);
        }
        acc4[ni] = a;
      }
    __syncthreads();   // everyone done with eb before its buffer is re-staged
  }

  // --- finalize: combine the two j half-groups, normalize, store ---
#pragma unroll
  for (int ni = 0; ni < 4; ++ni) {
    float o = acc4[ni] + __shfl_xor(acc4[ni], 16);
    o /= l_run;
    if (g == 0) O[(size_t)bi * INNER_ + wave * DH_ + ni * 16 + L0] = o;
  }
}

// ---------------------------------------------------------------------------
// Kernel 3: out = O(640x512) @ Wo(512x256) + bo.   grid (40,4) block 128
// ---------------------------------------------------------------------------
__global__ __launch_bounds__(128) void out_proj(
    const float* __restrict__ O, const float* __restrict__ Wo,
    const float* __restrict__ bo, float* __restrict__ out) {
  const int rb = blockIdx.x;
  const int cb = blockIdx.y;
  const int wave = threadIdx.x >> 5;
  const int lane = threadIdx.x & 31;
  const int L0 = lane & 15, g = lane >> 4;
  const int row0 = rb * 16;
  const int col0 = cb * 64 + wave * 16;

  v8f c = {0.f, 0.f, 0.f, 0.f, 0.f, 0.f, 0.f, 0.f};
#pragma unroll
  for (int k0 = 0; k0 < INNER_; k0 += 32) {
    const float* ap = O + (size_t)(row0 + L0) * INNER_ + k0 + 16 * g;
    v16bf a;
#pragma unroll
    for (int e = 0; e < 8; ++e) { a[e] = (__bf16)ap[e]; a[e + 8] = (__bf16)ap[8 + e]; }
    const float* bp = Wo + (size_t)(k0 + 16 * g) * DIM_ + col0 + L0;
    v16bf bb;
#pragma unroll
    for (int e = 0; e < 16; ++e) bb[e] = (__bf16)bp[e * DIM_];
    c = __builtin_amdgcn_wmma_f32_16x16x32_bf16(false, a, false, bb, (short)0, c,
                                                false, false);
  }
  const float bv = bo[col0 + L0];
#pragma unroll
  for (int r = 0; r < 8; ++r)
    out[(size_t)(row0 + r + 8 * g) * DIM_ + col0 + L0] = c[r] + bv;
}

// ---------------------------------------------------------------------------
extern "C" void kernel_launch(void* const* d_in, const int* in_sizes, int n_in,
                              void* d_out, int out_size, void* d_ws, size_t ws_size,
                              hipStream_t stream) {
  (void)in_sizes; (void)n_in; (void)out_size; (void)ws_size;
  const float* nodes = (const float*)d_in[0];
  const float* edges = (const float*)d_in[1];
  /* d_in[2] = mask: all-true per setup, unused */
  const float* Wq  = (const float*)d_in[3];
  const float* bq  = (const float*)d_in[4];
  const float* Wkv = (const float*)d_in[5];
  const float* bkv = (const float*)d_in[6];
  const float* We  = (const float*)d_in[7];
  const float* be  = (const float*)d_in[8];
  const float* Wo  = (const float*)d_in[9];
  const float* bo  = (const float*)d_in[10];

  float* ws = (float*)d_ws;
  const size_t sz = (size_t)B_ * N_ * INNER_;   // 327680 floats
  float*  Q   = ws;
  float*  Kt  = ws + sz;
  float*  Vt  = ws + 2 * sz;
  float*  O   = ws + 3 * sz;
  __bf16* Wet = (__bf16*)(ws + 4 * sz);         // 512x128 bf16 (128 KB)

  transpose_we<<<dim3(256), 256, 0, stream>>>(We, Wet);
  proj_gemm<INNER_, false><<<dim3(40, 8), 128, 0, stream>>>(nodes, Wq, bq, 0, Q);
  proj_gemm<2 * INNER_, true><<<dim3(40, 8), 128, 0, stream>>>(nodes, Wkv, bkv, 0, Kt);
  proj_gemm<2 * INNER_, true><<<dim3(40, 8), 128, 0, stream>>>(nodes, Wkv, bkv, INNER_, Vt);
  fused_attn<<<dim3(B_ * N_), 256, 0, stream>>>(edges, Wet, be, Q, Kt, Vt, O);
  out_proj<<<dim3(40, 4), 128, 0, stream>>>(O, Wo, bo, (float*)d_out);
}
